// MultiHeadAttentionV2_15015205667432
// MI455X (gfx1250) — compile-verified, matrix operand order checked by reading, not run on hardware
//
#include <hip/hip_runtime.h>

// ---------------- problem constants ----------------
#define D_IN   1024
#define D_OUT  1024
#define NH     16
#define HD     64          // head dim
#define SEQ    2048
#define BATCH  2

typedef __attribute__((ext_vector_type(16))) _Float16 v16h;
typedef __attribute__((ext_vector_type(8)))  _Float16 v8h;
typedef __attribute__((ext_vector_type(8)))  float    v8f;

// ---------------- WMMA helpers ----------------
__device__ static inline v8f wmma_f16(v16h a, v16h b, v8f c) {
  // D(16x16 f32) = A(16x32 f16) x B(32x16 f16) + C
  return __builtin_amdgcn_wmma_f32_16x16x32_f16(
      /*neg_a=*/false, a, /*neg_b=*/false, b,
      /*c_mod=*/(short)0, c, /*reuse_a=*/false, /*reuse_b=*/false);
}

__device__ static inline v16h mk16(v8h lo, v8h hi) {
  v16h r;
#pragma unroll
  for (int i = 0; i < 8; ++i) { r[i] = lo[i]; r[i + 8] = hi[i]; }
  return r;
}

// Load a 16x32 f16 A fragment or a 32x16 B fragment from an n-major /
// k-contiguous matrix: base = tile origin, ld = row stride (halfwords).
// ISA layout (wave32, 16-bit operands):
//   lanes 0-15 : row = lane,     k = {0..7, 16..23}
//   lanes 16-31: row = lane-16,  k = {8..15, 24..31}
__device__ static inline v16h load_frag(const _Float16* base, int ld, int lane) {
  const int r  = lane & 15;
  const int kb = (lane >> 4) << 3;            // 0 or 8
  const _Float16* p = base + (size_t)r * ld + kb;
  const v8h lo = *reinterpret_cast<const v8h*>(p);        // k = kb .. kb+7
  const v8h hi = *reinterpret_cast<const v8h*>(p + 16);   // k = kb+16 .. kb+23
  return mk16(lo, hi);
}

// 32x64 output tile: 8 WMMAs, B fragments reused across two row tiles.
__device__ static inline void mma8(v8f acc[8], v16h a0, v16h a1,
                                   v16h b0, v16h b1, v16h b2, v16h b3) {
  acc[0] = wmma_f16(a0, b0, acc[0]);
  acc[1] = wmma_f16(a0, b1, acc[1]);
  acc[2] = wmma_f16(a0, b2, acc[2]);
  acc[3] = wmma_f16(a0, b3, acc[3]);
  acc[4] = wmma_f16(a1, b0, acc[4]);
  acc[5] = wmma_f16(a1, b1, acc[5]);
  acc[6] = wmma_f16(a1, b2, acc[6]);
  acc[7] = wmma_f16(a1, b3, acc[7]);
}

// ---------------- conversion kernels ----------------
__global__ void f32_to_f16_kernel(const float* __restrict__ in,
                                  _Float16* __restrict__ out, int n) {
  int i = blockIdx.x * blockDim.x + threadIdx.x;
  if (i < n) out[i] = (_Float16)in[i];
}

// W[k][n] (f32, 1024x1024) -> Wt[n][k] (f16)
__global__ void transpose_to_f16_kernel(const float* __restrict__ W,
                                        _Float16* __restrict__ Wt) {
  int n = blockIdx.x * blockDim.x + threadIdx.x;
  int k = blockIdx.y;
  Wt[(size_t)n * D_IN + k] = (_Float16)W[(size_t)k * D_OUT + n];
}

// ---------------- QKV projection ----------------
// One wave: 32x64 tile of one of {Q,K,V}; k-loop software-pipelined with
// register double buffering. Output f16, head-major [b,h,s,d].
__global__ __launch_bounds__(32)
void qkv_gemm_kernel(const _Float16* __restrict__ xh,
                     const _Float16* __restrict__ wtq,
                     const _Float16* __restrict__ wtk,
                     const _Float16* __restrict__ wtv,
                     _Float16* __restrict__ Q,
                     _Float16* __restrict__ K,
                     _Float16* __restrict__ V) {
  const int lane  = threadIdx.x;
  const int m0    = blockIdx.x * 32;           // row tile over B*S
  const int head  = blockIdx.y;                // 64-wide col block = head
  const int which = blockIdx.z;                // 0=Q 1=K 2=V
  const _Float16* wt = (which == 0) ? wtq : (which == 1) ? wtk : wtv;
  _Float16* out      = (which == 0) ? Q   : (which == 1) ? K   : V;

  const _Float16* A0 = xh + (size_t)m0 * D_IN;
  const _Float16* A1 = A0 + (size_t)16 * D_IN;
  const _Float16* B0 = wt + (size_t)head * HD * D_IN;

  v8f acc[8] = {};
  // prologue (k = 0)
  v16h a0 = load_frag(A0, D_IN, lane);
  v16h a1 = load_frag(A1, D_IN, lane);
  v16h b0 = load_frag(B0 + (size_t)0 * 16 * D_IN, D_IN, lane);
  v16h b1 = load_frag(B0 + (size_t)1 * 16 * D_IN, D_IN, lane);
  v16h b2 = load_frag(B0 + (size_t)2 * 16 * D_IN, D_IN, lane);
  v16h b3 = load_frag(B0 + (size_t)3 * 16 * D_IN, D_IN, lane);

  for (int k0 = 32; k0 < D_IN; k0 += 32) {
    const v16h a0n = load_frag(A0 + k0, D_IN, lane);
    const v16h a1n = load_frag(A1 + k0, D_IN, lane);
    const v16h b0n = load_frag(B0 + (size_t)0 * 16 * D_IN + k0, D_IN, lane);
    const v16h b1n = load_frag(B0 + (size_t)1 * 16 * D_IN + k0, D_IN, lane);
    const v16h b2n = load_frag(B0 + (size_t)2 * 16 * D_IN + k0, D_IN, lane);
    const v16h b3n = load_frag(B0 + (size_t)3 * 16 * D_IN + k0, D_IN, lane);
    mma8(acc, a0, a1, b0, b1, b2, b3);
    a0 = a0n; a1 = a1n; b0 = b0n; b1 = b1n; b2 = b2n; b3 = b3n;
  }
  mma8(acc, a0, a1, b0, b1, b2, b3);   // epilogue

  const int rbase = (lane >> 4) * 8;
  const int col   = lane & 15;
#pragma unroll
  for (int mi = 0; mi < 2; ++mi)
#pragma unroll
    for (int j = 0; j < 4; ++j) {
      const int d = j * 16 + col;
#pragma unroll
      for (int r = 0; r < 8; ++r) {
        const int row = m0 + mi * 16 + rbase + r;     // = b*SEQ + s
        const int bb  = row >> 11;
        const int s   = row & (SEQ - 1);
        out[(((size_t)bb * NH + head) * SEQ + s) * HD + d] =
            (_Float16)acc[mi * 4 + j][r];
      }
    }
}

// ---------------- flash attention (cooperative) ----------------
// 128-thread block (4 waves) handles a 64-query band of one (b,h).
// K tile and transposed V tile staged ONCE per block in LDS; each wave
// runs online softmax for its own 16-query tile.
#define KLD 72   // LDS row stride, halfwords (144B: 16B-aligned, bank-spread)
__global__ __launch_bounds__(128)
void attn_kernel(const _Float16* __restrict__ Q,
                 const _Float16* __restrict__ K,
                 const _Float16* __restrict__ V,
                 _Float16* __restrict__ ctx) {
  __shared__ __align__(16) _Float16 k_lds[32 * KLD];       // [key][d]
  __shared__ __align__(16) _Float16 vt_lds[HD * KLD];      // [d][key]
  __shared__ __align__(16) _Float16 p_lds[4][16 * KLD];    // per-wave P tiles

  const int tid   = threadIdx.x;
  const int lane  = tid & 31;
  const int wave  = tid >> 5;
  const int qband = blockIdx.x * 64;
  const int q0    = qband + wave * 16;
  const int bh    = blockIdx.y;                 // b*NH + h
  const _Float16* Qp = Q + (size_t)bh * SEQ * HD;
  const _Float16* Kp = K + (size_t)bh * SEQ * HD;
  const _Float16* Vp = V + (size_t)bh * SEQ * HD;

  // Q fragments (head-dim k = 0..31, 32..63) held in registers.
  const v16h qa0 = load_frag(Qp + (size_t)q0 * HD +  0, HD, lane);
  const v16h qa1 = load_frag(Qp + (size_t)q0 * HD + 32, HD, lane);

  v8f o[4] = {};
  float m_i[8], l_i[8];
#pragma unroll
  for (int r = 0; r < 8; ++r) { m_i[r] = -1e30f; l_i[r] = 0.0f; }

  const int rbase = (lane >> 4) * 8;
  const int col   = lane & 15;
  const float scale = 0.125f;                   // 1/sqrt(64)
  const int q0u = __builtin_amdgcn_readfirstlane(q0);   // scalar for uniform branch
  const int nkb = (qband >> 5) + 2;             // keys <= qband+63

  for (int kb = 0; kb < nkb; ++kb) {
    const int key0 = kb * 32;

    // ---- cooperative stage: K tile row-major [key][d]
    {
      const int krow = tid >> 2;                // 0..31
      const int coff = (tid & 3) * 16;          // halfword offset
      const _Float16* src = Kp + (size_t)(key0 + krow) * HD + coff;
      const v8h d0 = *reinterpret_cast<const v8h*>(src);
      const v8h d1 = *reinterpret_cast<const v8h*>(src + 8);
      *reinterpret_cast<v8h*>(&k_lds[krow * KLD + coff])     = d0;
      *reinterpret_cast<v8h*>(&k_lds[krow * KLD + coff + 8]) = d1;
    }
    // ---- cooperative stage: V tile transposed [d][key]
    {
      const int key   = tid & 31;
      const int dbase = (tid >> 5) * 16;
      const _Float16* src = Vp + (size_t)(key0 + key) * HD + dbase;
      const v8h e0 = *reinterpret_cast<const v8h*>(src);
      const v8h e1 = *reinterpret_cast<const v8h*>(src + 8);
#pragma unroll
      for (int e = 0; e < 8; ++e) {
        vt_lds[(dbase + e) * KLD + key]     = e0[e];
        vt_lds[(dbase + 8 + e) * KLD + key] = e1[e];
      }
    }
    __syncthreads();

    if (key0 <= q0u + 15) {    // scalar, wave-uniform: EXEC stays all-ones
      // ---- scores: two 16-key tiles from LDS K tile
      v8f s0 = {}, s1 = {};
      s0 = wmma_f16(qa0, load_frag(k_lds, KLD, lane), s0);
      s0 = wmma_f16(qa1, load_frag(k_lds + 32, KLD, lane), s0);
      s1 = wmma_f16(qa0, load_frag(k_lds + 16 * KLD, KLD, lane), s1);
      s1 = wmma_f16(qa1, load_frag(k_lds + 16 * KLD + 32, KLD, lane), s1);

      // ---- scale + causal mask, per-row max
      float rmax[8];
#pragma unroll
      for (int r = 0; r < 8; ++r) {
        const int q = q0 + rbase + r;
        float a = s0[r] * scale; if (key0 + col      > q) a = -1e30f; s0[r] = a;
        float b = s1[r] * scale; if (key0 + 16 + col > q) b = -1e30f; s1[r] = b;
        rmax[r] = fmaxf(a, b);
      }
#pragma unroll
      for (int r = 0; r < 8; ++r)
#pragma unroll
        for (int mask = 8; mask >= 1; mask >>= 1)
          rmax[r] = fmaxf(rmax[r], __shfl_xor(rmax[r], mask, 32));

      // ---- online softmax update + rescale accumulators
      float rsum[8];
#pragma unroll
      for (int r = 0; r < 8; ++r) {
        const float nm = fmaxf(m_i[r], rmax[r]);
        const float p0 = __expf(s0[r] - nm);
        const float p1 = __expf(s1[r] - nm);
        s0[r] = p0; s1[r] = p1;
        rsum[r] = p0 + p1;
        const float corr = __expf(m_i[r] - nm);
        m_i[r] = nm;
        l_i[r] *= corr;
#pragma unroll
        for (int j = 0; j < 4; ++j) o[j][r] *= corr;
      }
#pragma unroll
      for (int r = 0; r < 8; ++r) {
#pragma unroll
        for (int mask = 8; mask >= 1; mask >>= 1)
          rsum[r] += __shfl_xor(rsum[r], mask, 32);
        l_i[r] += rsum[r];
      }

      // ---- stage P (f16) to this wave's private LDS tile (DS in-order per wave)
      _Float16* pw = p_lds[wave];
#pragma unroll
      for (int r = 0; r < 8; ++r) {
        pw[(rbase + r) * KLD + col]      = (_Float16)s0[r];
        pw[(rbase + r) * KLD + 16 + col] = (_Float16)s1[r];
      }
      __builtin_amdgcn_wave_barrier();   // pin compiler ordering store->load

      // ---- ctx += P(16x32) x Vt(32x16) per 16-wide d block
      const v16h pf = load_frag(pw, KLD, lane);
#pragma unroll
      for (int j = 0; j < 4; ++j) {
        const v16h vf = load_frag(vt_lds + (size_t)(j * 16) * KLD, KLD, lane);
        o[j] = wmma_f16(pf, vf, o[j]);
      }
    }
    __syncthreads();   // all waves: protect K/V tiles before next staging
  }

  // ---- normalize, store ctx f16 [b,h,s,d]
#pragma unroll
  for (int j = 0; j < 4; ++j) {
    const int d = j * 16 + col;
#pragma unroll
    for (int r = 0; r < 8; ++r) {
      const float val = o[j][r] / l_i[r];
      const int q = q0 + rbase + r;
      ctx[(size_t)bh * SEQ * HD + (size_t)q * HD + d] = (_Float16)val;
    }
  }
}

// ---------------- output projection ----------------
// A fragment loader over head-major ctx: row = b*S+s, col k = h*64+d.
__device__ static inline v16h load_ctx_frag(const _Float16* ctx, int row0,
                                            int k0, int lane) {
  const int r  = lane & 15;
  const int kb = (lane >> 4) << 3;
  const int row = row0 + r;
  const int bb  = row >> 11;
  const int s   = row & (SEQ - 1);
  const _Float16* cbase = ctx + (size_t)bb * NH * SEQ * HD + (size_t)s * HD;
  const int klo = k0 + kb, khi = klo + 16;   // chunks stay within a head block
  const v8h lo = *reinterpret_cast<const v8h*>(
      cbase + (size_t)(klo >> 6) * SEQ * HD + (klo & 63));
  const v8h hi = *reinterpret_cast<const v8h*>(
      cbase + (size_t)(khi >> 6) * SEQ * HD + (khi & 63));
  return mk16(lo, hi);
}

__global__ __launch_bounds__(32)
void out_proj_kernel(const _Float16* __restrict__ ctx,
                     const _Float16* __restrict__ wto,
                     const float* __restrict__ bias,
                     float* __restrict__ out) {
  const int lane = threadIdx.x;
  const int m0 = blockIdx.x * 32;
  const int n0 = blockIdx.y * 64;
  const _Float16* B0 = wto + (size_t)n0 * D_OUT;

  v8f acc[8] = {};
  v16h a0 = load_ctx_frag(ctx, m0,      0, lane);
  v16h a1 = load_ctx_frag(ctx, m0 + 16, 0, lane);
  v16h b0 = load_frag(B0 + (size_t)0 * 16 * D_OUT, D_OUT, lane);
  v16h b1 = load_frag(B0 + (size_t)1 * 16 * D_OUT, D_OUT, lane);
  v16h b2 = load_frag(B0 + (size_t)2 * 16 * D_OUT, D_OUT, lane);
  v16h b3 = load_frag(B0 + (size_t)3 * 16 * D_OUT, D_OUT, lane);

  for (int k0 = 32; k0 < D_OUT; k0 += 32) {
    const v16h a0n = load_ctx_frag(ctx, m0,      k0, lane);
    const v16h a1n = load_ctx_frag(ctx, m0 + 16, k0, lane);
    const v16h b0n = load_frag(B0 + (size_t)0 * 16 * D_OUT + k0, D_OUT, lane);
    const v16h b1n = load_frag(B0 + (size_t)1 * 16 * D_OUT + k0, D_OUT, lane);
    const v16h b2n = load_frag(B0 + (size_t)2 * 16 * D_OUT + k0, D_OUT, lane);
    const v16h b3n = load_frag(B0 + (size_t)3 * 16 * D_OUT + k0, D_OUT, lane);
    mma8(acc, a0, a1, b0, b1, b2, b3);
    a0 = a0n; a1 = a1n; b0 = b0n; b1 = b1n; b2 = b2n; b3 = b3n;
  }
  mma8(acc, a0, a1, b0, b1, b2, b3);

  const int rbase = (lane >> 4) * 8;
  const int col   = lane & 15;
#pragma unroll
  for (int mi = 0; mi < 2; ++mi)
#pragma unroll
    for (int j = 0; j < 4; ++j) {
      const int n = n0 + j * 16 + col;
      const float bv = bias[n];
#pragma unroll
      for (int rr = 0; rr < 8; ++rr) {
        const int orow = m0 + mi * 16 + rbase + rr;
        out[(size_t)orow * D_OUT + n] = acc[mi * 4 + j][rr] + bv;
      }
    }
}

// ---------------- host launcher ----------------
extern "C" void kernel_launch(void* const* d_in, const int* in_sizes, int n_in,
                              void* d_out, int out_size, void* d_ws, size_t ws_size,
                              hipStream_t stream) {
  const float* x   = (const float*)d_in[0];
  const float* Wq  = (const float*)d_in[1];
  const float* Wk  = (const float*)d_in[2];
  const float* Wv  = (const float*)d_in[3];
  const float* Wo  = (const float*)d_in[4];
  const float* bo  = (const float*)d_in[5];
  float* out = (float*)d_out;

  char* ws = (char*)d_ws;
  const size_t XH_BYTES = (size_t)BATCH * SEQ * D_IN * sizeof(_Float16);   // 8 MB
  const size_t WT_BYTES = (size_t)D_IN * D_OUT * sizeof(_Float16);         // 2 MB
  _Float16* xh  = (_Float16*)ws;                 ws += XH_BYTES;
  _Float16* wtq = (_Float16*)ws;                 ws += WT_BYTES;
  _Float16* wtk = (_Float16*)ws;                 ws += WT_BYTES;
  _Float16* wtv = (_Float16*)ws;                 ws += WT_BYTES;
  _Float16* wto = (_Float16*)ws;                 ws += WT_BYTES;
  _Float16* Qb  = (_Float16*)ws;                 ws += XH_BYTES;
  _Float16* Kb  = (_Float16*)ws;                 ws += XH_BYTES;
  _Float16* Vb  = (_Float16*)ws;                 ws += XH_BYTES;
  _Float16* ctx = (_Float16*)ws;                 ws += XH_BYTES;   // 48 MB total

  // 1) down-convert inputs (x straight, weights transposed to [n][k] f16)
  {
    const int n = BATCH * SEQ * D_IN;
    f32_to_f16_kernel<<<(n + 255) / 256, 256, 0, stream>>>(x, xh, n);
    dim3 tg(D_OUT / 256, D_IN);
    transpose_to_f16_kernel<<<tg, 256, 0, stream>>>(Wq, wtq);
    transpose_to_f16_kernel<<<tg, 256, 0, stream>>>(Wk, wtk);
    transpose_to_f16_kernel<<<tg, 256, 0, stream>>>(Wv, wtv);
    transpose_to_f16_kernel<<<tg, 256, 0, stream>>>(Wo, wto);
  }

  // 2) Q/K/V projections (WMMA, 32x64 tiles, pipelined)
  qkv_gemm_kernel<<<dim3(BATCH * SEQ / 32, NH, 3), 32, 0, stream>>>(
      xh, wtq, wtk, wtv, Qb, Kb, Vb);

  // 3) causal flash attention (WMMA + online softmax, LDS-shared K/V tiles)
  attn_kernel<<<dim3(SEQ / 64, BATCH * NH), 128, 0, stream>>>(Qb, Kb, Vb, ctx);

  // 4) output projection + bias (WMMA, f32 out)
  out_proj_kernel<<<dim3(BATCH * SEQ / 32, D_OUT / 64), 32, 0, stream>>>(
      ctx, wto, bo, out);
}
